// VectorQuantizer_48687749267721
// MI455X (gfx1250) — matrix operand order repaired and checked by reading, not compiled, not run on hardware
//
#include <hip/hip_runtime.h>

// ---------------------------------------------------------------------------
// VQ-VAE codebook step for MI455X (gfx1250, wave32, WMMA).
// Dominant cost: 16384x8192x512 distance GEMM -> v_wmma_f32_16x16x32_f16.
// Codebook scaled by 1024 before f16 conversion to avoid f16 denormals
// (entries are uniform in +-1/8192).  argmin uses ||e'||^2 - 2*S*(z.e')
// which preserves ordering.
// ---------------------------------------------------------------------------

typedef __attribute__((ext_vector_type(16))) _Float16 v16h;
typedef __attribute__((ext_vector_type(8)))  _Float16 v8h;
typedef __attribute__((ext_vector_type(8)))  float    v8f;

#define N_CODES   8192
#define CODE_DIM  512
#define M_TOTAL   16384            // 16 * 1024
#define EMB_ELEMS (8192ull * 512ull)
#define Z_ELEMS   (16384ull * 512ull)
#define S_SCALE   1024.0f

// ---------------------------------------------------------------------------
// K1: embedding f32 -> f16 (scaled) + per-code sum of squares of the
//     quantized values.  One block per code.
// ---------------------------------------------------------------------------
__global__ __launch_bounds__(256) void vq_prep_emb(
    const float* __restrict__ emb, _Float16* __restrict__ E16,
    float* __restrict__ sumsq)
{
    __shared__ float red[256];
    const int n = blockIdx.x;
    const float*   er = emb + (size_t)n * CODE_DIM;
    _Float16*      eo = E16 + (size_t)n * CODE_DIM;
    float s = 0.0f;
    for (int d = threadIdx.x; d < CODE_DIM; d += 256) {
        _Float16 q = (_Float16)(er[d] * S_SCALE);
        eo[d] = q;
        float qf = (float)q;
        s += qf * qf;
    }
    red[threadIdx.x] = s;
    __syncthreads();
    for (int st = 128; st > 0; st >>= 1) {
        if (threadIdx.x < st) red[threadIdx.x] += red[threadIdx.x + st];
        __syncthreads();
    }
    if (threadIdx.x == 0) sumsq[n] = red[0];
}

// ---------------------------------------------------------------------------
// K2: z f32 -> f16 (unscaled; z ~ N(0,1) is safely in f16 normal range).
// ---------------------------------------------------------------------------
__global__ __launch_bounds__(256) void vq_prep_z(
    const float* __restrict__ z, _Float16* __restrict__ Z16)
{
    for (size_t i = (size_t)blockIdx.x * blockDim.x + threadIdx.x;
         i < Z_ELEMS; i += (size_t)gridDim.x * blockDim.x)
        Z16[i] = (_Float16)z[i];
}

// ---------------------------------------------------------------------------
// K3: init accumulators: new_ema_w = ema_w * DECAY ; counts = 0 ; loss = 0.
// ---------------------------------------------------------------------------
__global__ __launch_bounds__(256) void vq_init(
    const float* __restrict__ ema_w, float* __restrict__ new_ema_w,
    float* __restrict__ counts, float* __restrict__ loss_acc)
{
    for (size_t i = (size_t)blockIdx.x * blockDim.x + threadIdx.x;
         i < EMB_ELEMS; i += (size_t)gridDim.x * blockDim.x)
        new_ema_w[i] = ema_w[i] * 0.99f;
    if (blockIdx.x == 0) {
        for (int n = threadIdx.x; n < N_CODES; n += blockDim.x) counts[n] = 0.0f;
        if (threadIdx.x == 0) *loss_acc = 0.0f;
    }
}

// ---------------------------------------------------------------------------
// K4: the big one.  Distance GEMM + running argmin.
// Workgroup = 256 threads = 8 waves.  Block tile: 32 rows of z.
//   wave w: msub = w&1 selects 16-row subtile, ngrp = w>>1 selects n-column
//   group; per n-iteration each wave computes 4 16x16 WMMA tiles, K=512
//   fully accumulated (16 x v_wmma_f32_16x16x32_f16 per tile).
//
// Fragment layouts use the same contiguous K order for BOTH A and B:
//   A (lane L, half i): row = L%16, memK = 16*(L/16) + i  -> contiguous 32B
//   B (lane L, half i): col = L%16, memK = 16*(L/16) + i  -> contiguous 32B
// ---------------------------------------------------------------------------
__global__ __launch_bounds__(256) void vq_gemm_argmin(
    const _Float16* __restrict__ Z16, const _Float16* __restrict__ E16,
    const float* __restrict__ sumsq, int* __restrict__ idx_out,
    float* __restrict__ idx_f_out)
{
    __shared__ float lds_min[32 * 4];
    __shared__ int   lds_idx[32 * 4];

    const int tid  = threadIdx.x;
    const int w    = tid >> 5;
    const int L    = tid & 31;
    const int h2   = L >> 4;      // half of wave
    const int ln   = L & 15;      // lane within half
    const int msub = w & 1;
    const int ngrp = w >> 1;
    const int m0   = blockIdx.x * 32;
    const int mrow = m0 + msub * 16 + ln;

    const _Float16* zrow = Z16 + (size_t)mrow * CODE_DIM;

    float runmin[8];
    int   runidx[8];
#pragma unroll
    for (int r = 0; r < 8; ++r) { runmin[r] = 3.4e38f; runidx[r] = 0; }

    for (int niter = 0; niter < 32; ++niter) {
        const int nbase = niter * 256 + ngrp * 64;
        if (niter + 1 < 32)   // gfx1250 global_prefetch_b8 for next code chunk
            __builtin_prefetch(E16 + (size_t)(nbase + 256) * CODE_DIM, 0, 3);

        v8f acc[4];
#pragma unroll
        for (int nt = 0; nt < 4; ++nt)
#pragma unroll
            for (int i = 0; i < 8; ++i) acc[nt][i] = 0.0f;

        for (int kb = 0; kb < 16; ++kb) {
            // A fragment: 32 contiguous bytes of the z row
            v16h a = *(const v16h*)(zrow + kb * 32 + h2 * 16);
#pragma unroll
            for (int nt = 0; nt < 4; ++nt) {
                const _Float16* erow =
                    E16 + (size_t)(nbase + nt * 16 + ln) * CODE_DIM + kb * 32;
                v16h b = *(const v16h*)(erow + h2 * 16);
                acc[nt] = __builtin_amdgcn_wmma_f32_16x16x32_f16(
                    /*neg_a=*/false, a, /*neg_b=*/false, b,
                    /*c_mod=*/(short)0, acc[nt],
                    /*reuse_a=*/false, /*reuse_b=*/false);
            }
        }

        // score = ||e'||^2 - 2*S*(z.e) = sumsq - 2*acc*S ; lower = closer
#pragma unroll
        for (int nt = 0; nt < 4; ++nt) {
            const int n = nbase + nt * 16 + ln;
            const float ss = sumsq[n];
#pragma unroll
            for (int r = 0; r < 8; ++r) {
                float s = ss - (2.0f * S_SCALE) * acc[nt][r];
                if (s < runmin[r]) { runmin[r] = s; runidx[r] = n; }
            }
        }
    }

    // C/D layout: lane L, reg r  <->  (row = 8*(L/16)+r, col = L%16).
    // Reduce over the 16 columns held within each half-wave, tie -> low idx.
#pragma unroll
    for (int r = 0; r < 8; ++r) {
        float mv = runmin[r];
        int   mi = runidx[r];
#pragma unroll
        for (int off = 1; off < 16; off <<= 1) {
            float ov = __shfl_xor(mv, off, 32);
            int   oi = __shfl_xor(mi, off, 32);
            if (ov < mv || (ov == mv && oi < mi)) { mv = ov; mi = oi; }
        }
        if (ln == 0) {
            const int lm = msub * 16 + h2 * 8 + r;   // 0..31 local row
            lds_min[lm * 4 + ngrp] = mv;
            lds_idx[lm * 4 + ngrp] = mi;
        }
    }
    __syncthreads();

    if (tid < 32) {
        float bv = lds_min[tid * 4];
        int   bi = lds_idx[tid * 4];
#pragma unroll
        for (int g = 1; g < 4; ++g) {
            float v = lds_min[tid * 4 + g];
            int   i2 = lds_idx[tid * 4 + g];
            if (v < bv || (v == bv && i2 < bi)) { bv = v; bi = i2; }
        }
        idx_out[m0 + tid]  = bi;
        idx_f_out[m0 + tid] = (float)bi;
    }
}

// ---------------------------------------------------------------------------
// K5: gather z_q (= z_q_st numerically), loss partial, counts histogram,
//     dw scatter-add folded directly into new_ema_w (+= 0.01 * z).
//     One block per z row.
// ---------------------------------------------------------------------------
__global__ __launch_bounds__(256) void vq_scatter(
    const float* __restrict__ z, const float* __restrict__ emb,
    const int* __restrict__ idx, float* __restrict__ zq_out,
    float* __restrict__ new_ema_w, float* __restrict__ counts,
    float* __restrict__ loss_acc)
{
    __shared__ float red[256];
    const int m = blockIdx.x;
    const int c = idx[m];
    const float* zr = z   + (size_t)m * CODE_DIM;
    const float* er = emb + (size_t)c * CODE_DIM;
    float*       nw = new_ema_w + (size_t)c * CODE_DIM;
    float*       zo = zq_out    + (size_t)m * CODE_DIM;

    float lsum = 0.0f;
    for (int d = threadIdx.x; d < CODE_DIM; d += 256) {
        float zv = zr[d];
        float ev = er[d];
        zo[d] = ev;                       // z_q_st == z_q numerically
        float df = zv - ev;
        lsum += df * df;
        atomicAdd(&nw[d], 0.01f * zv);    // (1-DECAY) * dw
    }
    red[threadIdx.x] = lsum;
    __syncthreads();
    for (int st = 128; st > 0; st >>= 1) {
        if (threadIdx.x < st) red[threadIdx.x] += red[threadIdx.x + st];
        __syncthreads();
    }
    if (threadIdx.x == 0) {
        atomicAdd(loss_acc, red[0]);
        atomicAdd(&counts[c], 1.0f);
    }
}

// ---------------------------------------------------------------------------
// K6: EMA cluster size + laplace smoothing + perplexity + vq_loss.
//     Single workgroup.
// ---------------------------------------------------------------------------
__global__ __launch_bounds__(256) void vq_finalize(
    const float* __restrict__ ema_cs, const float* __restrict__ counts,
    const float* __restrict__ loss_acc, float* __restrict__ out_cs,
    float* __restrict__ out_loss, float* __restrict__ out_perp)
{
    __shared__ float rsum[256], rent[256];
    float nsum = 0.0f, ent = 0.0f;
    for (int n = threadIdx.x; n < N_CODES; n += 256) {
        float pre = ema_cs[n] * 0.99f + 0.01f * counts[n];
        nsum += pre;
        float p = counts[n] * (1.0f / (float)M_TOTAL);
        ent += p * logf(p + 1e-10f);
    }
    rsum[threadIdx.x] = nsum;
    rent[threadIdx.x] = ent;
    __syncthreads();
    for (int st = 128; st > 0; st >>= 1) {
        if (threadIdx.x < st) {
            rsum[threadIdx.x] += rsum[threadIdx.x + st];
            rent[threadIdx.x] += rent[threadIdx.x + st];
        }
        __syncthreads();
    }
    const float N = rsum[0];
    const float denom = 1.0f / (N + (float)N_CODES * 1e-5f);
    for (int n = threadIdx.x; n < N_CODES; n += 256) {
        float pre = ema_cs[n] * 0.99f + 0.01f * counts[n];
        out_cs[n] = (pre + 1e-5f) * denom * N;
    }
    if (threadIdx.x == 0) {
        *out_perp = expf(-rent[0]);
        // commitment == codebook numerically -> (COST + 1) * MSE
        *out_loss = 1.25f * loss_acc[0] * (1.0f / (float)(16384ull * 512ull));
    }
}

// ---------------------------------------------------------------------------
// K7: new_embedding = new_ema_w / new_cs[:, None]
// ---------------------------------------------------------------------------
__global__ __launch_bounds__(256) void vq_newemb(
    const float* __restrict__ new_ema_w, const float* __restrict__ new_cs,
    float* __restrict__ out_emb)
{
    for (size_t i = (size_t)blockIdx.x * blockDim.x + threadIdx.x;
         i < EMB_ELEMS; i += (size_t)gridDim.x * blockDim.x)
        out_emb[i] = new_ema_w[i] / new_cs[i >> 9];   // 512 = 1<<9
}

// ---------------------------------------------------------------------------
extern "C" void kernel_launch(void* const* d_in, const int* in_sizes, int n_in,
                              void* d_out, int out_size, void* d_ws, size_t ws_size,
                              hipStream_t stream)
{
    const float* z      = (const float*)d_in[0];   // [16,1024,512]
    const float* emb    = (const float*)d_in[1];   // [8192,512]
    const float* ema_cs = (const float*)d_in[2];   // [8192]
    const float* ema_w  = (const float*)d_in[3];   // [8192,512]

    float* out      = (float*)d_out;
    float* out_zq   = out;                 // 8,388,608
    float* out_idx  = out + 8388608;       //    16,384
    float* out_loss = out + 8404992;       //         1
    float* out_perp = out + 8404993;       //         1
    float* out_emb  = out + 8404994;       // 4,194,304
    float* out_cs   = out + 12599298;      //     8,192
    float* out_emaw = out + 12607490;      // 4,194,304

    char* ws = (char*)d_ws;
    _Float16* Z16   = (_Float16*)(ws);              // 16,777,216 B
    _Float16* E16   = (_Float16*)(ws + 16777216);   //  8,388,608 B
    float*    sumsq = (float*)(ws + 25165824);      //     32,768 B
    int*      idxb  = (int*)  (ws + 25198592);      //     65,536 B
    float*    cnts  = (float*)(ws + 25264128);      //     32,768 B
    float*    lossa = (float*)(ws + 25296896);      //          4 B

    vq_prep_emb   <<<N_CODES, 256, 0, stream>>>(emb, E16, sumsq);
    vq_prep_z     <<<4096,    256, 0, stream>>>(z, Z16);
    vq_init       <<<2048,    256, 0, stream>>>(ema_w, out_emaw, cnts, lossa);
    vq_gemm_argmin<<<M_TOTAL / 32, 256, 0, stream>>>(Z16, E16, sumsq, idxb, out_idx);
    vq_scatter    <<<M_TOTAL, 256, 0, stream>>>(z, emb, idxb, out_zq, out_emaw, cnts, lossa);
    vq_finalize   <<<1,       256, 0, stream>>>(ema_cs, cnts, lossa, out_cs, out_loss, out_perp);
    vq_newemb     <<<4096,    256, 0, stream>>>(out_emaw, out_cs, out_emb);
}
